// RouterToMeGlue_68994354643294
// MI455X (gfx1250) — matrix operand order repaired and checked by reading, not compiled, not run on hardware
//
#include <hip/hip_runtime.h>
#include <hip/hip_bf16.h>

typedef __attribute__((ext_vector_type(4)))  _Float16 v4h;
typedef __attribute__((ext_vector_type(8)))  _Float16 v8h;
typedef __attribute__((ext_vector_type(16))) _Float16 v16h;
typedef __attribute__((ext_vector_type(8)))  float    v8f;

constexpr int B_  = 128;
constexpr int L_  = 1024;
constexpr int D_  = 768;
constexpr int T2_ = 512;   // tokens per parity (L/2)

// ---- GEMM tiling ----
constexpr int M_TILE = 64;
constexpr int N_TILE = 128;
constexpr int K_TILE = 32;
constexpr int LDH    = 40;   // padded LDS tile stride (halves)
constexpr int SC_LD  = 132;  // padded score stride (floats)

// ---------------------------------------------------------------------------
// Kernel 1: per-token inverse L2 norm over D (float4 loads -> global_load_b128)
// ---------------------------------------------------------------------------
__global__ void __launch_bounds__(256)
tome_norm_kernel(const float* __restrict__ hs, float* __restrict__ invn) {
  const int tok = blockIdx.x;                 // 0 .. B*L-1
  const float4* row = (const float4*)(hs + (size_t)tok * D_);
  float s = 0.f;
  for (int q = threadIdx.x; q < D_ / 4; q += 256) {
    float4 v = row[q];
    s += v.x * v.x + v.y * v.y + v.z * v.z + v.w * v.w;
  }
  __shared__ float red[256];
  red[threadIdx.x] = s;
  __syncthreads();
  for (int off = 128; off > 0; off >>= 1) {
    if (threadIdx.x < off) red[threadIdx.x] += red[threadIdx.x + off];
    __syncthreads();
  }
  if (threadIdx.x == 0) invn[tok] = 1.0f / sqrtf(red[0]);
}

// ---------------------------------------------------------------------------
// Kernel 2: WMMA similarity GEMM fused with per-row max/argmax.
// One workgroup = one batch x one 64-row M-tile; loops four 128-col N-tiles.
// ---------------------------------------------------------------------------
__global__ void __launch_bounds__(256)
tome_gemm_argmax_kernel(const float* __restrict__ hs,
                        const float* __restrict__ invn,
                        float* __restrict__ node_max,
                        int*   __restrict__ node_idx) {
  constexpr int SMEM_TILES  = (M_TILE + N_TILE) * LDH * 2;   // halves->bytes
  constexpr int SMEM_SCORES = M_TILE * SC_LD * 4;
  constexpr int SMEM_BYTES  = (SMEM_SCORES > SMEM_TILES) ? SMEM_SCORES : SMEM_TILES;
  __shared__ __align__(16) unsigned char smem[SMEM_BYTES];
  __shared__ float pval[256];
  __shared__ int   pidx[256];

  _Float16* At = (_Float16*)smem;           // [M_TILE][LDH]
  _Float16* Bt = At + M_TILE * LDH;         // [N_TILE][LDH]
  float*    Sc = (float*)smem;              // [M_TILE][SC_LD] (aliases tiles)

  const int tid   = threadIdx.x;
  const int lane  = tid & 31;
  const int wave  = tid >> 5;
  const int waveM = wave >> 2;              // 0..1  (32 rows each)
  const int waveN = wave & 3;               // 0..3  (32 cols each)
  const int b     = blockIdx.x >> 3;
  const int mBase = (blockIdx.x & 7) * M_TILE;

  const size_t hsB  = (size_t)b * L_ * D_;
  const float* invB = invn + (size_t)b * L_;

  const int m16 = lane & 15;
  const int h   = lane >> 4;

  float best = -__builtin_inff();           // running best (rows owned by tid<64)
  int bestIdx = 0;

  for (int ntile = 0; ntile < 4; ++ntile) {
    const int nBase = ntile * N_TILE;
    v8f acc[2][2] = {};

    for (int k0 = 0; k0 < D_; k0 += K_TILE) {
      // A tile: even tokens, unit-norm scaled, fp32 float4 -> packed 4xf16
      #pragma unroll
      for (int e = 0; e < (M_TILE * K_TILE / 4) / 256; ++e) {
        int lin = tid + e * 256;
        int row = lin >> 3, col = (lin & 7) * 4;
        int tok = 2 * (mBase + row);
        float4 vv = *(const float4*)(hs + hsB + (size_t)tok * D_ + k0 + col);
        float sc = invB[tok];
        v4h h4;
        h4[0] = (_Float16)(vv.x * sc); h4[1] = (_Float16)(vv.y * sc);
        h4[2] = (_Float16)(vv.z * sc); h4[3] = (_Float16)(vv.w * sc);
        *(v4h*)(At + row * LDH + col) = h4;
      }
      // B tile: odd tokens
      #pragma unroll
      for (int e = 0; e < (N_TILE * K_TILE / 4) / 256; ++e) {
        int lin = tid + e * 256;
        int row = lin >> 3, col = (lin & 7) * 4;
        int tok = 2 * (nBase + row) + 1;
        float4 vv = *(const float4*)(hs + hsB + (size_t)tok * D_ + k0 + col);
        float sc = invB[tok];
        v4h h4;
        h4[0] = (_Float16)(vv.x * sc); h4[1] = (_Float16)(vv.y * sc);
        h4[2] = (_Float16)(vv.z * sc); h4[3] = (_Float16)(vv.w * sc);
        *(v4h*)(Bt + row * LDH + col) = h4;
      }
      __syncthreads();

      // A fragments: ISA 16-bit A 16x32 layout
      //   lane half h=0 -> K {0..7, 16..23}; h=1 -> K {8..15, 24..31}
      v16h aF[2], bF[2];
      #pragma unroll
      for (int mi = 0; mi < 2; ++mi) {
        const _Float16* p = At + (waveM * 32 + mi * 16 + m16) * LDH;
        v8h lo = *(const v8h*)(p + h * 8);
        v8h hi = *(const v8h*)(p + 16 + h * 8);
        aF[mi] = __builtin_shufflevector(lo, hi, 0,1,2,3,4,5,6,7,8,9,10,11,12,13,14,15);
      }
      // B fragments: column n = lane&15, same K order as A: h=0 -> K {0..7, 16..23};
      // h=1 -> K {8..15, 24..31}. Loading row-major b rows realizes B = b^T (a . b^T).
      #pragma unroll
      for (int ni = 0; ni < 2; ++ni) {
        const _Float16* p = Bt + (waveN * 32 + ni * 16 + m16) * LDH;
        v8h lo = *(const v8h*)(p + h * 8);
        v8h hi = *(const v8h*)(p + 16 + h * 8);
        bF[ni] = __builtin_shufflevector(lo, hi, 0,1,2,3,4,5,6,7,8,9,10,11,12,13,14,15);
      }
      #pragma unroll
      for (int mi = 0; mi < 2; ++mi)
        #pragma unroll
        for (int ni = 0; ni < 2; ++ni)
          acc[mi][ni] = __builtin_amdgcn_wmma_f32_16x16x32_f16(
              false, aF[mi], false, bF[ni], (short)0, acc[mi][ni], false, false);
      __syncthreads();
    }

    // Stage scores to LDS (C/D layout: VGPR r holds M = r + 8*(lane>>4),
    // N = lane&15).
    #pragma unroll
    for (int mi = 0; mi < 2; ++mi)
      #pragma unroll
      for (int ni = 0; ni < 2; ++ni)
        #pragma unroll
        for (int rr = 0; rr < 8; ++rr) {
          int row = waveM * 32 + mi * 16 + h * 8 + rr;
          int col = waveN * 32 + ni * 16 + m16;
          Sc[row * SC_LD + col] = acc[mi][ni][rr];
        }
    __syncthreads();

    // Parallel argmax: 4 threads per row scan 32 cols each, then ordered
    // combine (ascending column parts + strict '>' == jnp.argmax first-max).
    {
      int row = tid >> 2, part = tid & 3;
      float lb = -__builtin_inff(); int li = 0;
      int c0 = part * 32;
      for (int c = c0; c < c0 + 32; ++c) {
        float v = Sc[row * SC_LD + c];
        if (v > lb) { lb = v; li = nBase + c; }
      }
      pval[tid] = lb; pidx[tid] = li;
    }
    __syncthreads();
    if (tid < M_TILE) {
      #pragma unroll
      for (int p = 0; p < 4; ++p) {
        float v = pval[tid * 4 + p];
        if (v > best) { best = v; bestIdx = pidx[tid * 4 + p]; }
      }
    }
    __syncthreads();
  }

  if (tid < M_TILE) {
    int i = mBase + tid;
    bool zr = (i == 0);  // reference forces row 0 to -inf / argmax 0
    node_max[(size_t)b * T2_ + i] = zr ? -__builtin_inff() : best;
    node_idx[(size_t)b * T2_ + i] = zr ? 0 : bestIdx;
  }
}

// ---------------------------------------------------------------------------
// Kernel 3: per-batch bitonic sort (value desc, idx asc == stable argsort of
// -node_max); emits src_idx, dst_idx, compacted ascending unm_idx.
// ---------------------------------------------------------------------------
__global__ void __launch_bounds__(256)
tome_sort_kernel(const float* __restrict__ node_max,
                 const int*   __restrict__ node_idx,
                 int* __restrict__ src_idx, int* __restrict__ dst_idx,
                 int* __restrict__ unm_idx, int r) {
  const int b = blockIdx.x;
  __shared__ float sval[T2_];
  __shared__ int   sidx[T2_];
  __shared__ int   keep[T2_];
  for (int p = threadIdx.x; p < T2_; p += 256) {
    sval[p] = node_max[(size_t)b * T2_ + p];
    sidx[p] = p;
    keep[p] = 0;
  }
  __syncthreads();
  for (int k = 2; k <= T2_; k <<= 1) {
    for (int j = k >> 1; j > 0; j >>= 1) {
      for (int t = threadIdx.x; t < T2_; t += 256) {
        int ixj = t ^ j;
        if (ixj > t) {
          float v0 = sval[t], v1 = sval[ixj];
          int   i0 = sidx[t], i1 = sidx[ixj];
          bool inOrder = (v0 > v1) || (v0 == v1 && i0 < i1);
          bool asc = ((t & k) == 0);
          if (asc != inOrder) {
            sval[t] = v1; sval[ixj] = v0;
            sidx[t] = i1; sidx[ixj] = i0;
          }
        }
      }
      __syncthreads();
    }
  }
  for (int p = threadIdx.x; p < T2_; p += 256) {
    if (p < r) {
      int si = sidx[p];
      src_idx[b * T2_ + p] = si;
      dst_idx[b * T2_ + p] = node_idx[(size_t)b * T2_ + si];
    } else {
      keep[sidx[p]] = 1;
    }
  }
  __syncthreads();
  if (threadIdx.x == 0) {
    int c = 0;
    for (int i = 0; i < T2_; ++i)
      if (keep[i]) unm_idx[b * T2_ + (c++)] = i;
  }
}

// ---------------------------------------------------------------------------
// Kernel 4: seed output dst region with odd tokens (float4), counts = 1
// ---------------------------------------------------------------------------
__global__ void __launch_bounds__(256)
tome_init_dst_kernel(const float* __restrict__ hs, float* __restrict__ out_tok,
                     float* __restrict__ counts, int T, int u) {
  const int g = blockIdx.x, b = g >> 9, j = g & (T2_ - 1);
  const float4* src = (const float4*)(hs + ((size_t)b * L_ + 2 * j + 1) * D_);
  float4* dstp = (float4*)(out_tok + ((size_t)b * T + u + j) * D_);
  for (int q = threadIdx.x; q < D_ / 4; q += 256) dstp[q] = src[q];
  if (threadIdx.x == 0) counts[b * T2_ + j] = 1.0f;
}

// ---------------------------------------------------------------------------
// Kernel 5: scatter-add merged even tokens into their dst rows
// ---------------------------------------------------------------------------
__global__ void __launch_bounds__(256)
tome_scatter_kernel(const float* __restrict__ hs,
                    const int* __restrict__ src_idx, const int* __restrict__ dst_idx,
                    float* __restrict__ out_tok, float* __restrict__ counts,
                    int T, int u, int r) {
  const int g = blockIdx.x, b = g / r, s = g - b * r;
  const int si = src_idx[b * T2_ + s];
  const int dj = dst_idx[b * T2_ + s];
  const float* srow = hs + ((size_t)b * L_ + 2 * si) * D_;
  float* drow = out_tok + ((size_t)b * T + u + dj) * D_;
  for (int q = threadIdx.x; q < D_ / 4; q += 256) {
    float4 v = ((const float4*)srow)[q];
    atomicAdd(&drow[q * 4 + 0], v.x);
    atomicAdd(&drow[q * 4 + 1], v.y);
    atomicAdd(&drow[q * 4 + 2], v.z);
    atomicAdd(&drow[q * 4 + 3], v.w);
  }
  if (threadIdx.x == 0) atomicAdd(&counts[b * T2_ + dj], 1.0f);
}

// ---------------------------------------------------------------------------
// Kernel 6: write unmerged rows, divide dst rows by counts (float4)
// ---------------------------------------------------------------------------
__global__ void __launch_bounds__(256)
tome_finalize_kernel(const float* __restrict__ hs, const int* __restrict__ unm_idx,
                     const float* __restrict__ counts, float* __restrict__ out_tok,
                     int T, int u) {
  const int g = blockIdx.x, b = g / T, p = g - b * T;
  float4* orow = (float4*)(out_tok + ((size_t)b * T + p) * D_);
  if (p < u) {
    int si = unm_idx[b * T2_ + p];
    const float4* srow = (const float4*)(hs + ((size_t)b * L_ + 2 * si) * D_);
    for (int q = threadIdx.x; q < D_ / 4; q += 256) orow[q] = srow[q];
  } else {
    int j = p - u;
    float inv = 1.0f / counts[b * T2_ + j];
    for (int q = threadIdx.x; q < D_ / 4; q += 256) {
      float4 v = orow[q];
      v.x *= inv; v.y *= inv; v.z *= inv; v.w *= inv;
      orow[q] = v;
    }
  }
}

// ---------------------------------------------------------------------------
// Kernel 7: zero the output attention mask
// ---------------------------------------------------------------------------
__global__ void __launch_bounds__(256)
tome_mask_kernel(float* __restrict__ mask, int n) {
  int i = blockIdx.x * 256 + threadIdx.x;
  if (i < n) mask[i] = 0.f;
}

// ---------------------------------------------------------------------------
extern "C" void kernel_launch(void* const* d_in, const int* in_sizes, int n_in,
                              void* d_out, int out_size, void* d_ws, size_t ws_size,
                              hipStream_t stream) {
  (void)in_sizes; (void)n_in; (void)ws_size;
  const float* hs = (const float*)d_in[0];   // (B, L, D) fp32
  // d_in[1] = attention_mask (all zeros, unused); d_in[2] = K (device scalar,
  // unused: r is recovered from out_size below, keeping launch host-deterministic)
  float* out = (float*)d_out;

  int T = out_size / (B_ * (D_ + 1));        // preserved tokens per batch
  if (T < T2_) T = T2_;
  if (T > L_)  T = L_;
  const int u = T - T2_;                     // unmerged even tokens kept
  const int r = T2_ - u;                     // merged even tokens

  float* out_tok  = out;
  float* out_mask = out + (size_t)B_ * T * D_;

  // workspace layout (~2.1 MB)
  float* invn     = (float*)d_ws;
  float* node_max = invn + (size_t)B_ * L_;
  int*   node_idx = (int*)(node_max + (size_t)B_ * T2_);
  int*   src_idx  = node_idx + (size_t)B_ * T2_;
  int*   dst_idx  = src_idx + (size_t)B_ * T2_;
  int*   unm_idx  = dst_idx + (size_t)B_ * T2_;
  float* counts   = (float*)(unm_idx + (size_t)B_ * T2_);

  tome_norm_kernel<<<B_ * L_, 256, 0, stream>>>(hs, invn);
  tome_gemm_argmax_kernel<<<B_ * 8, 256, 0, stream>>>(hs, invn, node_max, node_idx);
  tome_sort_kernel<<<B_, 256, 0, stream>>>(node_max, node_idx, src_idx, dst_idx, unm_idx, r);
  tome_init_dst_kernel<<<B_ * T2_, 256, 0, stream>>>(hs, out_tok, counts, T, u);
  if (r > 0)
    tome_scatter_kernel<<<B_ * r, 256, 0, stream>>>(hs, src_idx, dst_idx, out_tok, counts, T, u, r);
  tome_finalize_kernel<<<B_ * T, 256, 0, stream>>>(hs, unm_idx, counts, out_tok, T, u);
  const int maskN = B_ * T;
  tome_mask_kernel<<<(maskN + 255) / 256, 256, 0, stream>>>(out_mask, maskN);
}
